// AttentionalGRU_50259707297877
// MI455X (gfx1250) — matrix operand order repaired and checked by reading, compiled, not run on hardware
//
#include <hip/hip_runtime.h>
#include <hip/hip_bf16.h>

#define B_SZ 4096
#define S_SZ 64
#define H_SZ 512
#define HH   (H_SZ * H_SZ)   // 262144
#define ROWS 32              // batch rows per workgroup
#define NTHR 512             // 16 waves

typedef __attribute__((ext_vector_type(16))) __bf16 v16bf;
typedef __attribute__((ext_vector_type(8)))  __bf16 v8bf;
typedef __attribute__((ext_vector_type(4)))  __bf16 v4bf;
typedef __attribute__((ext_vector_type(8)))  float  v8f;
typedef __attribute__((ext_vector_type(4)))  float  v4f;

__device__ __forceinline__ v8f wmma_bf16(v16bf a, v16bf b, v8f c) {
  return __builtin_amdgcn_wmma_f32_16x16x32_bf16(
      /*neg_a=*/false, a, /*neg_b=*/false, b,
      /*c_mod=*/(short)0, c, /*reuse_a=*/false, /*reuse_b=*/false);
}

// A-matrix 16x32 bf16 fragment from an LDS tile (row-major, stride 512).
// Contiguous k order (matches B): lane<16 -> M=lane, halfs = K{kb+0..15};
//             lane>=16 -> M=lane-16, halfs = K{kb+16..31}.
__device__ __forceinline__ v16bf load_a_tile(const __bf16* T, int kb, int lane) {
  const int m  = lane & 15;
  const int hi = (lane >> 4) << 4;            // 0 or 16
  const __bf16* p = T + m * H_SZ + kb + hi;
  v8bf lo  = *(const v8bf*)(p);               // ds_load_b128
  v8bf hi8 = *(const v8bf*)(p + 8);           // ds_load_b128
  return __builtin_shufflevector(lo, hi8, 0,1,2,3,4,5,6,7,8,9,10,11,12,13,14,15);
}

// B-matrix 32x16 bf16 fragment = Wt^T tile: B[k][n] = Wt[nb+n][kb+k].
// Contiguous k order (matches A): lane<16 -> N=lane, K=kb+0..15; lane>=16 -> N=lane-16, K=kb+16..31.
__device__ __forceinline__ v16bf load_b_tile(const __bf16* __restrict__ Wt,
                                             int nb, int kb, int lane) {
  const int n  = lane & 15;
  const int kh = (lane >> 4) << 4;            // 0 or 16
  return *(const v16bf*)(Wt + (size_t)(nb + n) * H_SZ + kb + kh);
}

__device__ __forceinline__ void stage_facts(const float* __restrict__ facts,
                                            __bf16* Fbf, int bbase, int t, int tid) {
  // ROWS x 512 f32 -> bf16 LDS tile; NTHR threads, coalesced float4 loads.
  #pragma unroll
  for (int j = 0; j < 8; ++j) {
    const int idx = j * (NTHR * 4) + tid * 4;   // 0..16383
    const int r = idx >> 9, h = idx & 511;
    v4f v = *(const v4f*)(facts + (size_t)(bbase + r) * S_SZ * H_SZ
                                 + (size_t)t * H_SZ + h);
    v4bf b;
    b[0] = (__bf16)v[0]; b[1] = (__bf16)v[1];
    b[2] = (__bf16)v[2]; b[3] = (__bf16)v[3];
    *(v4bf*)(&Fbf[idx]) = b;                    // ds_store_b64
  }
}

__global__ __launch_bounds__(256) void agru_prep(
    const float* __restrict__ Wr, const float* __restrict__ W,
    const float* __restrict__ Ur, const float* __restrict__ U,
    __bf16* __restrict__ o) {
  const int i = blockIdx.x * blockDim.x + threadIdx.x;   // 262144 total
  o[i]          = (__bf16)Wr[i];
  o[i +     HH] = (__bf16)W[i];
  o[i + 2 * HH] = (__bf16)Ur[i];
  o[i + 3 * HH] = (__bf16)U[i];
}

__global__ __launch_bounds__(NTHR) void agru_main(
    const float* __restrict__ facts, const float* __restrict__ G,
    const float* __restrict__ br, const float* __restrict__ bur,
    const float* __restrict__ bw, const float* __restrict__ bu,
    const __bf16* __restrict__ wts, float* __restrict__ out) {
  __shared__ __align__(16) __bf16 Cbf[ROWS * H_SZ];  // 32 KB (WMMA A operand)
  __shared__ __align__(16) __bf16 Fbf[ROWS * H_SZ];  // 32 KB (facts_t tile)

  const __bf16* WrB = wts;
  const __bf16* WB  = wts + HH;
  const __bf16* UrB = wts + 2 * HH;
  const __bf16* UB  = wts + 3 * HH;

  const int tid   = threadIdx.x;
  const int lane  = tid & 31;
  const int wave  = tid >> 5;                  // 0..15, owns 32 output columns
  const int bbase = blockIdx.x * ROWS;

  // C0 = 0 (bf16 shared copy)
  #pragma unroll
  for (int j = 0; j < 8; ++j) {
    const int idx = j * (NTHR * 4) + tid * 4;
    *(v4bf*)(&Cbf[idx]) = (v4bf)((__bf16)0.0f);
  }
  stage_facts(facts, Fbf, bbase, 0, tid);
  __syncthreads();

  const int nw    = wave * 32;                 // column base for this wave
  const int dcol  = lane & 15;                 // D-tile column within N-tile
  const int rbase = (lane >> 4) * 8;           // D-tile rows rbase..rbase+7

  // f32 carry state lives in registers: cc[mt][nt][r] = C[mt*16+rbase+r][nw+nt*16+dcol]
  float cc[2][2][8];
  #pragma unroll
  for (int mt = 0; mt < 2; ++mt)
    #pragma unroll
    for (int nt = 0; nt < 2; ++nt)
      #pragma unroll
      for (int r = 0; r < 8; ++r) cc[mt][nt][r] = 0.0f;

  // biases are t-invariant: hoist
  float brc[2], bwc[2], buc[2];
  #pragma unroll
  for (int nt = 0; nt < 2; ++nt) {
    const int col = nw + nt * 16 + dcol;
    brc[nt] = br[col] + bur[col];
    bwc[nt] = bw[col];
    buc[nt] = bu[col];
  }

  for (int t = 0; t < S_SZ; ++t) {
    float gv[2][8];
    #pragma unroll
    for (int mt = 0; mt < 2; ++mt)
      #pragma unroll
      for (int r = 0; r < 8; ++r)
        gv[mt][r] = G[(size_t)(bbase + mt * 16 + rbase + r) * S_SZ + t];

    // warm L2 for next step's facts tile while the GEMMs run
    if (t + 1 < S_SZ) {
      #pragma unroll
      for (int j = 0; j < 8; ++j) {
        const int idx = j * (NTHR * 4) + tid * 4;
        const int r = idx >> 9, h = idx & 511;
        __builtin_prefetch(facts + (size_t)(bbase + r) * S_SZ * H_SZ
                                 + (size_t)(t + 1) * H_SZ + h, 0, 3);
      }
    }

    // 16 accumulators: 4 matrices x 2 M-tiles x 2 N-tiles (128 VGPRs)
    v8f aWr[2][2], aW[2][2], aR[2][2], aU[2][2];
    #pragma unroll
    for (int mt = 0; mt < 2; ++mt)
      #pragma unroll
      for (int nt = 0; nt < 2; ++nt) {
        aWr[mt][nt] = (v8f)(0.0f); aW[mt][nt] = (v8f)(0.0f);
        aR[mt][nt]  = (v8f)(0.0f); aU[mt][nt] = (v8f)(0.0f);
      }

    // kk outer: each A pair feeds 8 WMMAs, each B fragment feeds 2 WMMAs.
    #pragma unroll 4
    for (int kk = 0; kk < 16; ++kk) {
      const int kb = kk * 32;
      v16bf af[2], ac[2];
      #pragma unroll
      for (int mt = 0; mt < 2; ++mt) {
        af[mt] = load_a_tile(Fbf + mt * 16 * H_SZ, kb, lane);
        ac[mt] = load_a_tile(Cbf + mt * 16 * H_SZ, kb, lane);
      }
      #pragma unroll
      for (int nt = 0; nt < 2; ++nt) {
        const int nb = nw + nt * 16;
        const v16bf bWr = load_b_tile(WrB, nb, kb, lane);
        const v16bf bW  = load_b_tile(WB,  nb, kb, lane);
        const v16bf bUr = load_b_tile(UrB, nb, kb, lane);
        const v16bf bU  = load_b_tile(UB,  nb, kb, lane);
        #pragma unroll
        for (int mt = 0; mt < 2; ++mt) {
          aWr[mt][nt] = wmma_bf16(af[mt], bWr, aWr[mt][nt]);
          aW[mt][nt]  = wmma_bf16(af[mt], bW,  aW[mt][nt]);
          aR[mt][nt]  = wmma_bf16(ac[mt], bUr, aR[mt][nt]);
          aU[mt][nt]  = wmma_bf16(ac[mt], bU,  aU[mt][nt]);
        }
      }
    }

    // elementwise GRU update; carry stays in registers
    #pragma unroll
    for (int mt = 0; mt < 2; ++mt)
      #pragma unroll
      for (int nt = 0; nt < 2; ++nt)
        #pragma unroll
        for (int r = 0; r < 8; ++r) {
          const float rpre = aWr[mt][nt][r] + aR[mt][nt][r] + brc[nt];
          const float sg   = 1.0f / (1.0f + __expf(-rpre));               // sigmoid
          const float hpre = aW[mt][nt][r] + bwc[nt] + sg * (aU[mt][nt][r] + buc[nt]);
          const float ht   = 2.0f / (1.0f + __expf(-2.0f * hpre)) - 1.0f; // tanh
          const float g    = gv[mt][r];
          cc[mt][nt][r] = g * ht + (1.0f - g) * cc[mt][nt][r];
        }
    __syncthreads();   // all waves done reading old Cbf / Fbf

    #pragma unroll
    for (int mt = 0; mt < 2; ++mt)
      #pragma unroll
      for (int nt = 0; nt < 2; ++nt) {
        const int col = nw + nt * 16 + dcol;
        #pragma unroll
        for (int r = 0; r < 8; ++r)
          Cbf[(mt * 16 + rbase + r) * H_SZ + col] = (__bf16)cc[mt][nt][r];
      }
    if (t + 1 < S_SZ) stage_facts(facts, Fbf, bbase, t + 1, tid);
    __syncthreads();   // new Cbf / next facts visible before next GEMM phase
  }

  // final C (B,H) f32 straight from registers
  #pragma unroll
  for (int mt = 0; mt < 2; ++mt)
    #pragma unroll
    for (int nt = 0; nt < 2; ++nt) {
      const int col = nw + nt * 16 + dcol;
      #pragma unroll
      for (int r = 0; r < 8; ++r)
        out[(size_t)(bbase + mt * 16 + rbase + r) * H_SZ + col] = cc[mt][nt][r];
    }
}

extern "C" void kernel_launch(void* const* d_in, const int* in_sizes, int n_in,
                              void* d_out, int out_size, void* d_ws, size_t ws_size,
                              hipStream_t stream) {
  const float* facts = (const float*)d_in[0];
  const float* G     = (const float*)d_in[1];
  const float* Wr    = (const float*)d_in[2];
  const float* br    = (const float*)d_in[3];
  const float* Ur    = (const float*)d_in[4];
  const float* bur   = (const float*)d_in[5];
  const float* W     = (const float*)d_in[6];
  const float* bw    = (const float*)d_in[7];
  const float* U     = (const float*)d_in[8];
  const float* bu    = (const float*)d_in[9];
  __bf16* wts = (__bf16*)d_ws;   // 4 x 512x512 bf16 = 2 MB scratch

  agru_prep<<<HH / 256, 256, 0, stream>>>(Wr, W, Ur, U, wts);
  agru_main<<<B_SZ / ROWS, NTHR, 0, stream>>>(facts, G, br, bur, bw, bu, wts,
                                              (float*)d_out);
}
